// LocalPointAttention_42202348650559
// MI455X (gfx1250) — matrix operand order repaired and checked by reading, not compiled, not run on hardware
//
#include <hip/hip_runtime.h>

// ---------------- static config (mirrors reference) ----------------
#define N_ATOMS   1024
#define GRID_N    33
#define G3        (GRID_N*GRID_N*GRID_N)   // 35937
#define C_LAT     64
#define P_ENC     4
#define C_A       32
#define C_TOT     224                       // 32 + 3*64
#define PIN       608                       // 8*(64+12)
#define N_HEAD    4
#define KEY_DIM   16
#define OUT_DIM   32
#define RADIUS2   0.01f

typedef __attribute__((ext_vector_type(16))) _Float16 v16h;
typedef __attribute__((ext_vector_type(8)))  float    v8f;

// ---------------- workspace layout (bytes, 32B-aligned regions) -----
#define O_DENS    0u
#define O_TDENS   143872u                    // 35937*4 -> pad
#define O_W16P    287744u                    // proj weights f16, B-fragment-swizzled, 608*64
#define O_W16QKV  365568u                    // 3 * 224*64 f16, B-fragment-swizzled
#define O_ALPHA   451584u                    // 1024*64 f32
#define O_CORR    713728u
#define O_A16     975872u                    // 1024*224 f16
#define O_Q       1434624u                   // 1024*64 f32 (scaled)
#define O_AK      1696768u
#define O_AV      1958912u

// ---------------- kernel 1: zero density grid ----------------
__global__ void k_zero(float* dens) {
    int i = blockIdx.x * blockDim.x + threadIdx.x;
    if (i < G3) dens[i] = 0.0f;
}

// ---------------- kernel 2: density scatter (encode) ----------------
__global__ void k_scatter(const float* __restrict__ pos, float* dens) {
    int i = blockIdx.x * blockDim.x + threadIdx.x;
    if (i >= N_ATOMS) return;
    int cx = (int)floorf(pos[i*3+0] * (GRID_N-1));
    int cy = (int)floorf(pos[i*3+1] * (GRID_N-1));
    int cz = (int)floorf(pos[i*3+2] * (GRID_N-1));
    cx = cx < 0 ? 0 : (cx > GRID_N-2 ? GRID_N-2 : cx);
    cy = cy < 0 ? 0 : (cy > GRID_N-2 ? GRID_N-2 : cy);
    cz = cz < 0 ? 0 : (cz > GRID_N-2 ? GRID_N-2 : cz);
    atomicAdd(dens + (cx*GRID_N + cy)*GRID_N + cz, 1.0f);
}

// ---------------- kernel 3: tanh ----------------
__global__ void k_tanh(const float* __restrict__ dens, float* tdens) {
    int i = blockIdx.x * blockDim.x + threadIdx.x;
    if (i < G3) tdens[i] = tanhf(dens[i]);
}

// ---------------- kernel 4: convert + swizzle weights into B-fragment order --
// Fragment order: frag[((kk*64 + n)*2 + hi)*16 + e] = W[(kk*32 + e + hi*16)*64 + n]
// so each lane's whole v16h B operand is one contiguous, 32B-aligned block.
__global__ void k_cvt(const float* __restrict__ pw, const float* __restrict__ wq,
                      const float* __restrict__ wk, const float* __restrict__ wv,
                      _Float16* w16p, _Float16* w16qkv) {
    int i = blockIdx.x * blockDim.x + threadIdx.x;
    const int NP = PIN * 64;            // 38912
    const int NQ = C_TOT * 64;          // 14336
    if (i < NP) {
        int e  = i & 15;
        int hi = (i >> 4) & 1;
        int n  = (i >> 5) & 63;
        int kk = i >> 11;
        int K  = kk*32 + e + hi*16;
        w16p[i] = (_Float16)pw[K*64 + n];
    } else if (i < NP + 3*NQ) {
        int r = i - NP;
        int sel = r / NQ, rr = r % NQ;
        int e  = rr & 15;
        int hi = (rr >> 4) & 1;
        int n  = (rr >> 5) & 63;
        int kk = rr >> 11;
        int K  = kk*32 + e + hi*16;
        const float* s = sel == 0 ? wq : (sel == 1 ? wk : wv);
        w16qkv[r] = (_Float16)s[K*64 + n];
    }
}

// ---------------- kernel 5: projection GEMM (WMMA) ----------------
// grid (64, 2): 16 atoms per block; y==0 -> latent feats, y==1 -> corr feats.
// feats[16][608] built in LDS, then [16x608]@[608x64] via v_wmma_f32_16x16x32_f16.
__global__ __launch_bounds__(128) void k_proj(
        const float* __restrict__ latent, const float* __restrict__ tdens,
        const float* __restrict__ encp,  const float* __restrict__ pos,
        const float* __restrict__ box,   const _Float16* __restrict__ w16p,
        float* outAlpha, float* outCorr) {
    __shared__ _Float16 fe[16][PIN];
    __shared__ float ppos[16][3];
    __shared__ int   pidx[16][3];
    const int tid = threadIdx.x;
    const int corrPass = blockIdx.y;

    if (tid < 16) {
        int i = blockIdx.x * 16 + tid;
        #pragma unroll
        for (int d = 0; d < 3; ++d) {
            float p = pos[i*3+d] * (float)(GRID_N-1) / box[d];
            ppos[tid][d] = p;
            pidx[tid][d] = (int)floorf(p);
        }
    }
    __syncthreads();

    for (int e0 = tid; e0 < 16 * PIN; e0 += 128) {
        int r = e0 / PIN, e = e0 % PIN;
        int corner = e / 76, q = e % 76;
        int ix = pidx[r][0] + ((corner >> 2) & 1);
        int iy = pidx[r][1] + ((corner >> 1) & 1);
        int iz = pidx[r][2] + (corner & 1);
        float val;
        if (q < 64) {
            int cx = ix > GRID_N-1 ? GRID_N-1 : (ix < 0 ? 0 : ix);
            int cy = iy > GRID_N-1 ? GRID_N-1 : (iy < 0 ? 0 : iy);
            int cz = iz > GRID_N-1 ? GRID_N-1 : (iz < 0 ? 0 : iz);
            int v = (cx*GRID_N + cy)*GRID_N + cz;
            val = corrPass ? tdens[v] * encp[q] : latent[v*C_LAT + q];
        } else {
            int p = q - 64, d = p >> 2, m = p & 3;
            float ind = (d == 0) ? (float)ix : (d == 1) ? (float)iy : (float)iz;
            float sd = (ind - ppos[r][d] + 1.0f) * 0.5f;
            val = cosf(sd * (float)(m + 1));
        }
        fe[r][e] = (_Float16)val;
    }
    __syncthreads();

    const int wv_ = tid >> 5, lane = tid & 31;
    const int n0 = wv_ * 16;
    const int mrow = lane & 15;
    const int hi = (lane >> 4) & 1;
    const v16h* bw = (const v16h*)w16p;    // swizzled fragments, 32B each
    v8f c = {};
    for (int kk = 0; kk < PIN / 32; ++kk) {        // 19 K-steps
        v16h a;
        int kbase = kk * 32;
        #pragma unroll
        for (int e = 0; e < 16; ++e) {
            int ka = kbase + e + (hi << 4);               // A 16x32 f16 lane layout (contiguous k = 16*hi + e)
            a[e] = fe[mrow][ka];
        }
        v16h b = bw[(kk*64 + (n0 + mrow))*2 + hi];        // 2x global_load_b128
        c = __builtin_amdgcn_wmma_f32_16x16x32_f16(false, a, false, b,
                                                   (short)0, c, false, false);
    }
    float* outp = corrPass ? outCorr : outAlpha;
    #pragma unroll
    for (int r = 0; r < 8; ++r) {
        int M = r + (hi << 3);
        outp[(blockIdx.x * 16 + M) * 64 + (n0 + mrow)] = c[r];
    }
}

// ---------------- kernel 6: assemble a = [rep|alpha|corr|corr-alpha] -> f16 ---
__global__ void k_assemble(const float* __restrict__ rep, const float* __restrict__ alpha,
                           const float* __restrict__ corr, _Float16* a16) {
    int i = blockIdx.x * blockDim.x + threadIdx.x;
    if (i >= N_ATOMS * C_TOT) return;
    int r = i / C_TOT, j = i % C_TOT;
    float v;
    if (j < 32)       v = rep[r*32 + j];
    else if (j < 96)  v = alpha[r*64 + (j-32)];
    else if (j < 160) v = corr[r*64 + (j-96)];
    else              v = corr[r*64 + (j-160)] - alpha[r*64 + (j-160)];
    a16[i] = (_Float16)v;
}

// ---------------- kernel 7: QKV GEMM (WMMA) ----------------
// grid (64, 3): [1024,224]@[224,64]; y selects Wq/Wk/Wv (q scaled by 0.25).
__global__ __launch_bounds__(128) void k_qkv(
        const _Float16* __restrict__ a16, const _Float16* __restrict__ w16qkv,
        float* qb, float* akb, float* avb) {
    __shared__ _Float16 at[16][C_TOT];
    const int tid = threadIdx.x;
    for (int e0 = tid; e0 < 16 * C_TOT; e0 += 128) {
        int r = e0 / C_TOT, e = e0 % C_TOT;
        at[r][e] = a16[(blockIdx.x * 16 + r) * C_TOT + e];
    }
    __syncthreads();

    const v16h* bw = (const v16h*)(w16qkv + blockIdx.y * (C_TOT * 64));
    const int wv_ = tid >> 5, lane = tid & 31;
    const int n0 = wv_ * 16;
    const int mrow = lane & 15;
    const int hi = (lane >> 4) & 1;
    v8f c = {};
    for (int kk = 0; kk < C_TOT / 32; ++kk) {      // 7 K-steps
        v16h a;
        int kbase = kk * 32;
        #pragma unroll
        for (int e = 0; e < 16; ++e) {
            int ka = kbase + e + (hi << 4);
            a[e] = at[mrow][ka];
        }
        v16h b = bw[(kk*64 + (n0 + mrow))*2 + hi];
        c = __builtin_amdgcn_wmma_f32_16x16x32_f16(false, a, false, b,
                                                   (short)0, c, false, false);
    }
    float* outp = blockIdx.y == 0 ? qb : (blockIdx.y == 1 ? akb : avb);
    float sc = blockIdx.y == 0 ? 0.25f : 1.0f;     // KEY_DIM^-0.5
    #pragma unroll
    for (int r = 0; r < 8; ++r) {
        int M = r + (hi << 3);
        outp[(blockIdx.x * 16 + M) * 64 + (n0 + mrow)] = c[r] * sc;
    }
}

// ---------------- kernel 8: fused neighbour search + attention ----------------
// 1 wave32 per atom, 4 atoms per 128-thread block. Brute force radius search
// (expected ~4.3 neighbours/atom), ballot-compacted list, 2-pass softmax.
__global__ __launch_bounds__(128) void k_attn(
        const float* __restrict__ pos, const float* __restrict__ qb,
        const float* __restrict__ akb, const float* __restrict__ avb,
        const float* __restrict__ Wo,  const float* __restrict__ bo,
        float* out) {
    __shared__ float sp[N_ATOMS * 3];
    __shared__ int   nl[4][512];
    __shared__ float ns[4][512];
    __shared__ __align__(16) float qsh[4][64];
    __shared__ float oacc[4][64];
    __shared__ float ssum[4][4];
    const int tid = threadIdx.x, w = tid >> 5, lane = tid & 31;
    const int b = blockIdx.x * 4 + w;

    for (int i = tid; i < N_ATOMS * 3; i += 128) sp[i] = pos[i];
    qsh[w][lane]      = qb[b*64 + lane];
    qsh[w][lane + 32] = qb[b*64 + lane + 32];
    oacc[w][lane] = 0.0f; oacc[w][lane + 32] = 0.0f;
    if (lane < 4) ssum[w][lane] = 0.0f;
    __syncthreads();

    const float bx = sp[b*3+0], by = sp[b*3+1], bz = sp[b*3+2];
    int cnt = 0;
    for (int c0 = 0; c0 < N_ATOMS / 32; ++c0) {
        int j = c0 * 32 + lane;
        float dx = bx - sp[j*3+0], dy = by - sp[j*3+1], dz = bz - sp[j*3+2];
        float d2 = dx*dx + dy*dy + dz*dz;
        bool within = d2 < RADIUS2;
        unsigned m32 = (unsigned)__ballot(within);
        if (within) {
            int p = cnt + __popc(m32 & ((1u << lane) - 1u));
            if (p < 512) {
                float nd = sqrtf(d2 + 1e-12f);
                if (nd < 1e-6f) nd += 1e9f;      // self-distance kill
                nl[w][p] = j;
                ns[w][p] = 1.0f / (nd * nd);     // inverse-square distance
            }
        }
        cnt += __popc(m32);
    }
    if (cnt > 512) cnt = 512;

    const float4* q4 = (const float4*)&qsh[w][0];

    // pass A: per-head running max of logits
    float m0 = -3.0e38f, m1 = -3.0e38f, m2 = -3.0e38f, m3 = -3.0e38f;
    for (int t = lane; t < cnt; t += 32) {
        int j = nl[w][t]; float s = ns[w][t];
        const float4* ak4 = (const float4*)(akb + j * 64);
        float d0 = 0, d1 = 0, d2 = 0, d3 = 0;
        #pragma unroll
        for (int cc = 0; cc < 4; ++cc) {          // 4x global_load_b128 each
            float4 a0 = ak4[cc],      qq0 = q4[cc];
            float4 a1 = ak4[4 + cc],  qq1 = q4[4 + cc];
            float4 a2 = ak4[8 + cc],  qq2 = q4[8 + cc];
            float4 a3 = ak4[12 + cc], qq3 = q4[12 + cc];
            d0 += a0.x*qq0.x + a0.y*qq0.y + a0.z*qq0.z + a0.w*qq0.w;
            d1 += a1.x*qq1.x + a1.y*qq1.y + a1.z*qq1.z + a1.w*qq1.w;
            d2 += a2.x*qq2.x + a2.y*qq2.y + a2.z*qq2.z + a2.w*qq2.w;
            d3 += a3.x*qq3.x + a3.y*qq3.y + a3.z*qq3.z + a3.w*qq3.w;
        }
        m0 = fmaxf(m0, s*d0); m1 = fmaxf(m1, s*d1);
        m2 = fmaxf(m2, s*d2); m3 = fmaxf(m3, s*d3);
    }
    #pragma unroll
    for (int off = 16; off >= 1; off >>= 1) {
        m0 = fmaxf(m0, __shfl_xor(m0, off, 32));
        m1 = fmaxf(m1, __shfl_xor(m1, off, 32));
        m2 = fmaxf(m2, __shfl_xor(m2, off, 32));
        m3 = fmaxf(m3, __shfl_xor(m3, off, 32));
    }

    // pass B: exp-sum + weighted value accumulation
    float o64[64];
    #pragma unroll
    for (int k = 0; k < 64; ++k) o64[k] = 0.0f;
    float s0 = 0, s1 = 0, s2 = 0, s3 = 0;
    for (int t = lane; t < cnt; t += 32) {
        int j = nl[w][t]; float s = ns[w][t];
        const float4* ak4 = (const float4*)(akb + j * 64);
        const float4* av4 = (const float4*)(avb + j * 64);
        float d0 = 0, d1 = 0, d2 = 0, d3 = 0;
        #pragma unroll
        for (int cc = 0; cc < 4; ++cc) {
            float4 a0 = ak4[cc],      qq0 = q4[cc];
            float4 a1 = ak4[4 + cc],  qq1 = q4[4 + cc];
            float4 a2 = ak4[8 + cc],  qq2 = q4[8 + cc];
            float4 a3 = ak4[12 + cc], qq3 = q4[12 + cc];
            d0 += a0.x*qq0.x + a0.y*qq0.y + a0.z*qq0.z + a0.w*qq0.w;
            d1 += a1.x*qq1.x + a1.y*qq1.y + a1.z*qq1.z + a1.w*qq1.w;
            d2 += a2.x*qq2.x + a2.y*qq2.y + a2.z*qq2.z + a2.w*qq2.w;
            d3 += a3.x*qq3.x + a3.y*qq3.y + a3.z*qq3.z + a3.w*qq3.w;
        }
        float e0 = __expf(s*d0 - m0), e1 = __expf(s*d1 - m1);
        float e2 = __expf(s*d2 - m2), e3 = __expf(s*d3 - m3);
        s0 += e0; s1 += e1; s2 += e2; s3 += e3;
        float w0 = e0*s, w1 = e1*s, w2 = e2*s, w3 = e3*s;
        #pragma unroll
        for (int cc = 0; cc < 4; ++cc) {
            float4 v0 = av4[cc], v1 = av4[4+cc], v2 = av4[8+cc], v3 = av4[12+cc];
            o64[cc*4+0]      += w0*v0.x; o64[cc*4+1]      += w0*v0.y;
            o64[cc*4+2]      += w0*v0.z; o64[cc*4+3]      += w0*v0.w;
            o64[16+cc*4+0]   += w1*v1.x; o64[16+cc*4+1]   += w1*v1.y;
            o64[16+cc*4+2]   += w1*v1.z; o64[16+cc*4+3]   += w1*v1.w;
            o64[32+cc*4+0]   += w2*v2.x; o64[32+cc*4+1]   += w2*v2.y;
            o64[32+cc*4+2]   += w2*v2.z; o64[32+cc*4+3]   += w2*v2.w;
            o64[48+cc*4+0]   += w3*v3.x; o64[48+cc*4+1]   += w3*v3.y;
            o64[48+cc*4+2]   += w3*v3.z; o64[48+cc*4+3]   += w3*v3.w;
        }
    }
    atomicAdd(&ssum[w][0], s0); atomicAdd(&ssum[w][1], s1);
    atomicAdd(&ssum[w][2], s2); atomicAdd(&ssum[w][3], s3);
    #pragma unroll
    for (int k = 0; k < 64; ++k) atomicAdd(&oacc[w][k], o64[k]);
    __syncthreads();

    // out = (o/sum) @ Wo + bo ; one output channel per lane
    float val = bo[lane];
    #pragma unroll
    for (int k = 0; k < 64; ++k)
        val += (oacc[w][k] / ssum[w][k >> 4]) * Wo[k * OUT_DIM + lane];
    out[b * OUT_DIM + lane] = val;
}

// ---------------- launcher ----------------
extern "C" void kernel_launch(void* const* d_in, const int* in_sizes, int n_in,
                              void* d_out, int out_size, void* d_ws, size_t ws_size,
                              hipStream_t stream) {
    (void)in_sizes; (void)n_in; (void)out_size; (void)ws_size;
    const float* encp = (const float*)d_in[0];
    const float* rep  = (const float*)d_in[1];
    const float* lat  = (const float*)d_in[2];
    const float* pos  = (const float*)d_in[3];
    const float* box  = (const float*)d_in[4];
    const float* pw   = (const float*)d_in[5];
    const float* Wq   = (const float*)d_in[6];
    const float* Wk   = (const float*)d_in[7];
    const float* Wv   = (const float*)d_in[8];
    const float* Wo   = (const float*)d_in[9];
    const float* bo   = (const float*)d_in[10];
    float* out = (float*)d_out;
    char* ws = (char*)d_ws;

    float*    dens    = (float*)   (ws + O_DENS);
    float*    tdens   = (float*)   (ws + O_TDENS);
    _Float16* w16p    = (_Float16*)(ws + O_W16P);
    _Float16* w16qkv  = (_Float16*)(ws + O_W16QKV);
    float*    alpha   = (float*)   (ws + O_ALPHA);
    float*    corr    = (float*)   (ws + O_CORR);
    _Float16* a16     = (_Float16*)(ws + O_A16);
    float*    qbuf    = (float*)   (ws + O_Q);
    float*    akbuf   = (float*)   (ws + O_AK);
    float*    avbuf   = (float*)   (ws + O_AV);

    k_zero   <<<(G3 + 255) / 256, 256, 0, stream>>>(dens);
    k_scatter<<<(N_ATOMS + 255) / 256, 256, 0, stream>>>(pos, dens);
    k_tanh   <<<(G3 + 255) / 256, 256, 0, stream>>>(dens, tdens);
    k_cvt    <<<(PIN*64 + 3*C_TOT*64 + 255) / 256, 256, 0, stream>>>(pw, Wq, Wk, Wv, w16p, w16qkv);
    k_proj   <<<dim3(N_ATOMS/16, 2), 128, 0, stream>>>(lat, tdens, encp, pos, box, w16p, alpha, corr);
    k_assemble<<<(N_ATOMS*C_TOT + 255) / 256, 256, 0, stream>>>(rep, alpha, corr, a16);
    k_qkv    <<<dim3(N_ATOMS/16, 3), 128, 0, stream>>>(a16, w16qkv, qbuf, akbuf, avbuf);
    k_attn   <<<N_ATOMS/4, 128, 0, stream>>>(pos, qbuf, akbuf, avbuf, Wo, bo, out);
}
